// MultimodalSparseDeformableTransformerEncoderLayer_13872744366701
// MI455X (gfx1250) — compile-verified, matrix operand order checked by reading, not run on hardware
//
#include <hip/hip_runtime.h>
#include <cstdint>
#include <cstddef>

// ---------------------------------------------------------------------------
// Types for CDNA5 WMMA / async copies
// ---------------------------------------------------------------------------
typedef __attribute__((ext_vector_type(16))) _Float16 v16h;
typedef __attribute__((ext_vector_type(8)))  float    v8f;
typedef __attribute__((ext_vector_type(4)))  int      v4i;

// pointer-to-addrspace-qualified v4i, as required by the async-copy builtins
typedef __attribute__((address_space(1))) v4i* gv4i_p;   // global
typedef __attribute__((address_space(3))) v4i* sv4i_p;   // LDS

union Frag { v16h h; unsigned int u[8]; };

#define LN_EPSF 1e-5f

// gfx1250 async global->LDS copy path (ASYNCcnt), with sync fallback
#if defined(__HIP_DEVICE_COMPILE__) && \
    __has_builtin(__builtin_amdgcn_global_load_async_to_lds_b128)
#define USE_ASYNC_LDS 1
#else
#define USE_ASYNC_LDS 0
#endif

// Problem constants (match reference setup_inputs)
constexpr int kB  = 8;
constexpr int kS  = 3840;
constexpr int kD  = 256;
constexpr int kL  = 4;
constexpr int kH  = 8;
constexpr int kP  = 4;
constexpr int kM  = kB * kS;       // 30720 rows
constexpr int kHP = kH * kL * kP;  // 128

// ---------------------------------------------------------------------------
// WMMA GEMM, all-f16 dataflow: Out = act(X[M,K] @ WT[N,K]^T + bias)
// X is f16 row-major, WT is the weight pre-transposed to [N][K] f16 so both
// tiles stage into LDS as contiguous b128 copies (async-to-LDS capable).
// Block: 256 threads = 8 waves (2Mx4N); block tile 64x128; wave tile 32x32
// (2x2 WMMA, 4 independent v_wmma_f32_16x16x32_f16 per 32-wide k-step).
// ---------------------------------------------------------------------------
template <bool RELU, bool W32, bool W16>
__global__ __launch_bounds__(256) void gemm_h_wmma(
    const _Float16* __restrict__ X,   // [M,K] f16
    const _Float16* __restrict__ WT,  // [N,K] f16 (pre-transposed weight)
    const float* __restrict__ Bias,   // [N]
    float* __restrict__ Out32, _Float16* __restrict__ Out16,
    int M, int N, int K) {
  __shared__ _Float16 As[64][40];    // [m][k], 16B-aligned rows (80B stride)
  __shared__ _Float16 Bs[128][40];   // [n][k]

  const int tid  = threadIdx.x;
  const int lane = tid & 31;
  const int wave = tid >> 5;     // 0..7
  const int wm   = wave >> 2;    // 0..1  -> 32-row slab
  const int wn   = wave & 3;     // 0..3  -> 32-col slab
  const int m0   = blockIdx.y * 64;
  const int n0   = blockIdx.x * 128;

  const int r     = lane & 15;          // row (A) / col (B) within 16-tile
  const int khalf = (lane >> 4) << 3;   // upper half-wave holds K+8 block

  // cooperative staging coordinates (one b128 for A, two b128 for B)
  const int arow = tid >> 2;            // 0..63
  const int acol = (tid & 3) << 3;      // 0,8,16,24 halves
  const int brow = tid >> 1;            // 0..127
  const int bcol = (tid & 1) << 4;      // 0,16 halves (+8 for second b128)

  v8f acc00 = {}, acc01 = {}, acc10 = {}, acc11 = {};

  for (int k0 = 0; k0 < K; k0 += 32) {
    _Float16* xsrc = const_cast<_Float16*>(X + (size_t)(m0 + arow) * K + k0 + acol);
    _Float16* bsrc = const_cast<_Float16*>(WT + (size_t)(n0 + brow) * K + k0 + bcol);
#if !USE_ASYNC_LDS
    const uint4 av = *(const uint4*)xsrc;
    const uint4 b0 = *(const uint4*)bsrc;
    const uint4 b1 = *(const uint4*)(bsrc + 8);
#endif
    __syncthreads();  // previous iteration's LDS reads complete
#if USE_ASYNC_LDS
    __builtin_amdgcn_global_load_async_to_lds_b128(
        (gv4i_p)(void*)xsrc,
        (sv4i_p)(void*)&As[arow][acol], 0, 0);
    __builtin_amdgcn_global_load_async_to_lds_b128(
        (gv4i_p)(void*)bsrc,
        (sv4i_p)(void*)&Bs[brow][bcol], 0, 0);
    __builtin_amdgcn_global_load_async_to_lds_b128(
        (gv4i_p)(void*)(bsrc + 8),
        (sv4i_p)(void*)&Bs[brow][bcol + 8], 0, 0);
    asm volatile("s_wait_asynccnt 0" ::: "memory");
#else
    *(uint4*)&As[arow][acol]     = av;
    *(uint4*)&Bs[brow][bcol]     = b0;
    *(uint4*)&Bs[brow][bcol + 8] = b1;
#endif
    __syncthreads();

    if (k0 + 32 < K) {  // hint next tiles toward cache (global_prefetch_b8)
      __builtin_prefetch(X + (size_t)(m0 + arow) * K + k0 + 32 + acol, 0, 1);
      __builtin_prefetch(WT + (size_t)(n0 + brow) * K + k0 + 32 + bcol, 0, 1);
    }

    // fragments: packed (k,k+1) dword reads per ISA 16-bit A/B layout
    Frag fa0, fa1, fb0, fb1;
#pragma unroll
    for (int j = 0; j < 8; ++j) {
      const int kk = ((j < 4) ? 0 : 16) + khalf + ((j & 3) << 1);
      fa0.u[j] = *(const unsigned int*)&As[wm * 32 + r][kk];
      fa1.u[j] = *(const unsigned int*)&As[wm * 32 + 16 + r][kk];
      fb0.u[j] = *(const unsigned int*)&Bs[wn * 32 + r][kk];
      fb1.u[j] = *(const unsigned int*)&Bs[wn * 32 + 16 + r][kk];
    }
    acc00 = __builtin_amdgcn_wmma_f32_16x16x32_f16(
        false, fa0.h, false, fb0.h, (short)0, acc00, false, false);
    acc01 = __builtin_amdgcn_wmma_f32_16x16x32_f16(
        false, fa0.h, false, fb1.h, (short)0, acc01, false, false);
    acc10 = __builtin_amdgcn_wmma_f32_16x16x32_f16(
        false, fa1.h, false, fb0.h, (short)0, acc10, false, false);
    acc11 = __builtin_amdgcn_wmma_f32_16x16x32_f16(
        false, fa1.h, false, fb1.h, (short)0, acc11, false, false);
  }

  // epilogue: C layout VGPR v -> M = v + 8*(lane>=16), N = lane&15
  const int obm = m0 + wm * 32 + ((lane >> 4) << 3);
  const int obn = n0 + wn * 32 + (lane & 15);
  const float bv0 = Bias[obn];
  const float bv1 = Bias[obn + 16];
#pragma unroll
  for (int v = 0; v < 8; ++v) {
    float v00 = acc00[v] + bv0;
    float v01 = acc01[v] + bv1;
    float v10 = acc10[v] + bv0;
    float v11 = acc11[v] + bv1;
    if (RELU) {
      v00 = fmaxf(v00, 0.0f); v01 = fmaxf(v01, 0.0f);
      v10 = fmaxf(v10, 0.0f); v11 = fmaxf(v11, 0.0f);
    }
    if (W32) {
      Out32[(size_t)(obm + v) * N + obn]           = v00;
      Out32[(size_t)(obm + v) * N + obn + 16]      = v01;
      Out32[(size_t)(obm + 16 + v) * N + obn]      = v10;
      Out32[(size_t)(obm + 16 + v) * N + obn + 16] = v11;
    }
    if (W16) {
      Out16[(size_t)(obm + v) * N + obn]           = (_Float16)v00;
      Out16[(size_t)(obm + v) * N + obn + 16]      = (_Float16)v01;
      Out16[(size_t)(obm + 16 + v) * N + obn]      = (_Float16)v10;
      Out16[(size_t)(obm + 16 + v) * N + obn + 16] = (_Float16)v11;
    }
  }
}

// ---------------------------------------------------------------------------
// Conversion / elementwise helpers
// ---------------------------------------------------------------------------
__global__ void cvt_f16_vec4(const float* __restrict__ in, _Float16* __restrict__ out,
                             int n4) {
  int i = blockIdx.x * blockDim.x + threadIdx.x;
  if (i >= n4) return;
  float4 x = ((const float4*)in)[i];
  _Float16 h[4] = {(_Float16)x.x, (_Float16)x.y, (_Float16)x.z, (_Float16)x.w};
  uint2 u;
  __builtin_memcpy(&u, h, 8);
  ((uint2*)out)[i] = u;
}

__global__ void add_f16_vec4(const float* __restrict__ a, const float* __restrict__ b,
                             _Float16* __restrict__ o, int n4) {
  int i = blockIdx.x * blockDim.x + threadIdx.x;
  if (i >= n4) return;
  float4 x = ((const float4*)a)[i];
  float4 y = ((const float4*)b)[i];
  _Float16 h[4] = {(_Float16)(x.x + y.x), (_Float16)(x.y + y.y),
                   (_Float16)(x.z + y.z), (_Float16)(x.w + y.w)};
  uint2 u;
  __builtin_memcpy(&u, h, 8);
  ((uint2*)o)[i] = u;
}

// WT[n*K + k] = (f16) W[k*N + n]
__global__ void transpose_cvt_w(const float* __restrict__ W, _Float16* __restrict__ WT,
                                int K, int N) {
  int i = blockIdx.x * blockDim.x + threadIdx.x;
  if (i >= K * N) return;
  const int k = i / N;
  const int n = i - k * N;
  WT[(size_t)n * K + k] = (_Float16)W[i];
}

// zero masked rows of the f16 value matrix (viewed as dwords, 128 per row)
__global__ void apply_mask_h(_Float16* __restrict__ v,
                             const unsigned char* __restrict__ mask, int ndw) {
  int i = blockIdx.x * blockDim.x + threadIdx.x;
  if (i >= ndw) return;
  if (mask[i >> 7]) ((unsigned int*)v)[i] = 0u;
}

// softmax over contiguous groups of 16 (L*P per head)
__global__ void softmax16(const float* __restrict__ in, float* __restrict__ out, int total) {
  int idx = blockIdx.x * blockDim.x + threadIdx.x;
  if (idx >= total) return;
  const int base = idx << 4;
  float m = -3.4e38f;
#pragma unroll
  for (int j = 0; j < 16; ++j) m = fmaxf(m, in[base + j]);
  float e[16];
  float s = 0.0f;
#pragma unroll
  for (int j = 0; j < 16; ++j) { e[j] = __expf(in[base + j] - m); s += e[j]; }
  const float inv = 1.0f / s;
#pragma unroll
  for (int j = 0; j < 16; ++j) out[base + j] = e[j] * inv;
}

// loc[m,h,l,p] = ref[m,l] + off[m,h,l,p] / ts[l]
__global__ void loc_kernel(const float* __restrict__ ref, const float* __restrict__ off,
                           const int* __restrict__ ts, float* __restrict__ loc, int n) {
  int i = blockIdx.x * blockDim.x + threadIdx.x;
  if (i >= n) return;
  const int m = i >> 7;
  const int r = i & 127;
  const int l = (r >> 2) & 3;
  loc[i] = ref[m * 4 + l] + off[i] / (float)ts[l];
}

// Deformable sampling: one wave per (m,h); lane = channel within Dh=32.
// V is f16 (halves gather bandwidth); accumulate f32; emit f16 for Wp GEMM.
__global__ __launch_bounds__(256) void msda_sample_h(
    const _Float16* __restrict__ V, const float* __restrict__ loc,
    const float* __restrict__ aw, const int* __restrict__ ts,
    const int* __restrict__ lsi, _Float16* __restrict__ out, int S) {
  const int gid  = blockIdx.x * (blockDim.x >> 5) + (threadIdx.x >> 5);
  const int lane = threadIdx.x & 31;
  const int m = gid >> 3;   // row (b*S + q)
  const int h = gid & 7;    // head
  const int b = m / S;
  const _Float16* vbase = V + (size_t)b * S * 256 + h * 32 + lane;
  const int base = m * 128 + h * 16;
  float acc = 0.0f;
#pragma unroll
  for (int l = 0; l < 4; ++l) {
    const int T  = ts[l];
    const int s0 = lsi[l];
#pragma unroll
    for (int p = 0; p < 4; ++p) {
      const int j = base + l * 4 + p;
      const float pos = loc[j] * (float)T - 0.5f;
      const float x0f = floorf(pos);
      const float w1  = pos - x0f;
      const int   xi  = (int)x0f;
      const float a   = aw[j];
      float v0 = 0.0f, v1 = 0.0f;
      if (xi >= 0 && xi < T)         v0 = (float)vbase[(size_t)(s0 + xi) * 256];
      if (xi + 1 >= 0 && xi + 1 < T) v1 = (float)vbase[(size_t)(s0 + xi + 1) * 256];
      acc += (v0 * (1.0f - w1) + v1 * w1) * a;
    }
  }
  out[(size_t)m * 256 + h * 32 + lane] = (_Float16)acc;
}

// out = LayerNorm(x + res) * g + b ; one wave per 256-wide row.
// Optionally also emits an f16 twin for downstream GEMM consumption.
template <bool W16>
__global__ __launch_bounds__(256) void ln_add(
    const float* __restrict__ x, const float* __restrict__ res,
    const float* __restrict__ g, const float* __restrict__ b,
    float* __restrict__ out, _Float16* __restrict__ out16) {
  const int row  = blockIdx.x * (blockDim.x >> 5) + (threadIdx.x >> 5);
  const int lane = threadIdx.x & 31;
  const float* px = x   + (size_t)row * 256;
  const float* pr = res + (size_t)row * 256;
  float v[8];
  float s = 0.0f;
#pragma unroll
  for (int j = 0; j < 8; ++j) {
    v[j] = px[lane + j * 32] + pr[lane + j * 32];
    s += v[j];
  }
#pragma unroll
  for (int off = 16; off > 0; off >>= 1) s += __shfl_xor(s, off, 32);
  const float mean = s * (1.0f / 256.0f);
  float t = 0.0f;
#pragma unroll
  for (int j = 0; j < 8; ++j) { const float d = v[j] - mean; t += d * d; }
#pragma unroll
  for (int off = 16; off > 0; off >>= 1) t += __shfl_xor(t, off, 32);
  const float inv = rsqrtf(t * (1.0f / 256.0f) + LN_EPSF);
#pragma unroll
  for (int j = 0; j < 8; ++j) {
    const int c = lane + j * 32;
    const float y = (v[j] - mean) * inv * g[c] + b[c];
    out[(size_t)row * 256 + c] = y;
    if (W16) out16[(size_t)row * 256 + c] = (_Float16)y;
  }
}

// ---------------------------------------------------------------------------
// Host-side GEMM dispatch
// ---------------------------------------------------------------------------
static void launch_gemm(bool relu, bool w32, bool w16,
                        const _Float16* X, const _Float16* WT, const float* bias,
                        float* o32, _Float16* o16, int N, int K, hipStream_t s) {
  dim3 grid(N / 128, kM / 64);
  if (relu)
    gemm_h_wmma<true, false, true><<<grid, 256, 0, s>>>(X, WT, bias, o32, o16, kM, N, K);
  else if (w32 && w16)
    gemm_h_wmma<false, true, true><<<grid, 256, 0, s>>>(X, WT, bias, o32, o16, kM, N, K);
  else if (w32)
    gemm_h_wmma<false, true, false><<<grid, 256, 0, s>>>(X, WT, bias, o32, o16, kM, N, K);
  else
    gemm_h_wmma<false, false, true><<<grid, 256, 0, s>>>(X, WT, bias, o32, o16, kM, N, K);
}

// ---------------------------------------------------------------------------
// Orchestration
// ---------------------------------------------------------------------------
extern "C" void kernel_launch(void* const* d_in, const int* in_sizes, int n_in,
                              void* d_out, int out_size, void* d_ws, size_t ws_size,
                              hipStream_t stream) {
  (void)in_sizes; (void)n_in; (void)out_size; (void)ws_size;

  const float* video_src = (const float*)d_in[0];
  const float* audio_src = (const float*)d_in[1];
  const float* video_pos = (const float*)d_in[2];
  const float* audio_pos = (const float*)d_in[3];
  const float* video_ref = (const float*)d_in[4];
  const float* audio_ref = (const float*)d_in[5];
  const int*   video_ts  = (const int*)d_in[6];
  const int*   video_lsi = (const int*)d_in[7];
  const unsigned char* video_mask = (const unsigned char*)d_in[8];
  const int*   audio_ts  = (const int*)d_in[9];
  const int*   audio_lsi = (const int*)d_in[10];
  const unsigned char* audio_mask = (const unsigned char*)d_in[11];
  const float* Wv = (const float*)d_in[12];
  const float* bv = (const float*)d_in[13];
  const float* Wo = (const float*)d_in[14];
  const float* bo = (const float*)d_in[15];
  const float* Wa = (const float*)d_in[16];
  const float* ba = (const float*)d_in[17];
  const float* Wp = (const float*)d_in[18];
  const float* bp = (const float*)d_in[19];
  const float* ln1_g = (const float*)d_in[20];
  const float* ln1_b = (const float*)d_in[21];
  const float* W1 = (const float*)d_in[22];
  const float* b1 = (const float*)d_in[23];
  const float* W2 = (const float*)d_in[24];
  const float* b2 = (const float*)d_in[25];
  const float* ln2_g = (const float*)d_in[26];
  const float* ln2_b = (const float*)d_in[27];

  float* outp = (float*)d_out;

  const size_t Mf = (size_t)kM * 256;   // 256-wide stride (elements)
  const size_t Ml = (size_t)kM * 128;   // 128-wide stride (elements)

  // ---- f32 workspace region
  float* wsf   = (float*)d_ws;
  float* TMP   = wsf + 0 * Mf;   // msda result (A/B) and FFN h2 scratch
  float* VIDEO = wsf + 1 * Mf;
  float* AUDIO = wsf + 2 * Mf;
  float* VAA   = wsf + 3 * Mf;
  float* AAV   = wsf + 4 * Mf;
  float* OFF   = wsf + 5 * Mf;          // [M,128]
  float* AW    = OFF + Ml;              // [M,128]
  float* LOC   = AW + Ml;               // [M,128]

  // ---- f16 workspace region (q16..SAMP16 contiguous so H16 can alias them)
  _Float16* wsh    = (_Float16*)(LOC + Ml);
  _Float16* q16    = wsh + 0 * Mf;
  _Float16* src16  = wsh + 1 * Mf;
  _Float16* V16    = wsh + 2 * Mf;
  _Float16* SAMP16 = wsh + 3 * Mf;
  _Float16* H16    = q16;               // spans 4*Mf = [M,1024] during FFN
  _Float16* VIDEO16 = wsh + 4 * Mf;
  _Float16* AUDIO16 = wsh + 5 * Mf;
  _Float16* VAA16   = wsh + 6 * Mf;
  _Float16* AAV16   = wsh + 7 * Mf;
  _Float16* WvT = wsh + 8 * Mf;          // [256,256]
  _Float16* WoT = WvT + 256 * 256;       // [128,256]
  _Float16* WaT = WoT + 128 * 256;       // [128,256]
  _Float16* WpT = WaT + 128 * 256;       // [256,256]
  _Float16* W1T = WpT + 256 * 256;       // [1024,256]
  _Float16* W2T = W1T + 1024 * 256;      // [256,1024]

  // d_out layout: aav, vaa, v_loc, v_aw, a_loc, a_aw
  float* o_aav  = outp;
  float* o_vaa  = outp + Mf;
  float* o_vloc = outp + 2 * Mf;
  float* o_vaw  = o_vloc + Ml;
  float* o_aloc = o_vaw + Ml;
  float* o_aaw  = o_aloc + Ml;

  const int nElem = kM * 256;
  const int n4    = nElem / 4;
  const int nOff  = kM * kHP;   // 3,932,160
  const int nSoft = kM * kH;    // 245,760
  const int nMask = kM * 128;   // dwords in f16 value matrix

  // ---- pre-transpose + convert all weights to f16 [N][K]
  auto tw = [&](const float* W, _Float16* WT, int K, int N) {
    const int n = K * N;
    transpose_cvt_w<<<(n + 255) / 256, 256, 0, stream>>>(W, WT, K, N);
  };
  tw(Wv, WvT, 256, 256);
  tw(Wo, WoT, 256, 128);
  tw(Wa, WaT, 256, 128);
  tw(Wp, WpT, 256, 256);
  tw(W1, W1T, 256, 1024);
  tw(W2, W2T, 1024, 256);

  auto msda_stage = [&](const _Float16* query16, const _Float16* srcv16,
                        const float* ref, const int* ts, const int* lsi,
                        const unsigned char* mask, float* loc_out, float* aw_out,
                        float* res32, _Float16* res16) {
    // value = src @ Wv + bv (f16 out), then zero masked rows
    launch_gemm(false, false, true, srcv16, WvT, bv, nullptr, V16, 256, 256, stream);
    apply_mask_h<<<(nMask + 255) / 256, 256, 0, stream>>>(V16, mask, nMask);
    // sampling offsets and attention scores (f32 out)
    launch_gemm(false, true, false, query16, WoT, bo, OFF, nullptr, 128, 256, stream);
    launch_gemm(false, true, false, query16, WaT, ba, aw_out, nullptr, 128, 256, stream);
    softmax16<<<(nSoft + 255) / 256, 256, 0, stream>>>(aw_out, aw_out, nSoft);
    loc_kernel<<<(nOff + 255) / 256, 256, 0, stream>>>(ref, OFF, ts, loc_out, nOff);
    // bilinear deformable sampling (f16 value in, f16 sampled out)
    msda_sample_h<<<nSoft / 8, 256, 0, stream>>>(V16, loc_out, aw_out, ts, lsi, SAMP16, kS);
    // output projection
    launch_gemm(false, true, res16 != nullptr, SAMP16, WpT, bp, res32, res16, 256, 256, stream);
  };

  auto ffn = [&](const float* x32, const _Float16* x16, float* out) {
    launch_gemm(true, false, true, x16, W1T, b1, nullptr, H16, 1024, 256, stream);
    launch_gemm(false, true, false, H16, W2T, b2, TMP, nullptr, 256, 1024, stream);
    ln_add<false><<<kM / 8, 256, 0, stream>>>(x32, TMP, ln2_g, ln2_b, out, nullptr);
  };

  // ---- Stage A: video self-attention -> VIDEO (+f16 twin)
  cvt_f16_vec4<<<(n4 + 255) / 256, 256, 0, stream>>>(video_src, src16, n4);
  add_f16_vec4<<<(n4 + 255) / 256, 256, 0, stream>>>(video_src, video_pos, q16, n4);
  msda_stage(q16, src16, video_ref, video_ts, video_lsi, video_mask, LOC, AW, TMP, nullptr);
  ln_add<true><<<kM / 8, 256, 0, stream>>>(video_src, TMP, ln1_g, ln1_b, VIDEO, VIDEO16);

  // ---- Stage B: audio self-attention -> AUDIO (+f16 twin)
  cvt_f16_vec4<<<(n4 + 255) / 256, 256, 0, stream>>>(audio_src, src16, n4);
  add_f16_vec4<<<(n4 + 255) / 256, 256, 0, stream>>>(audio_src, audio_pos, q16, n4);
  msda_stage(q16, src16, audio_ref, audio_ts, audio_lsi, audio_mask, LOC, AW, TMP, nullptr);
  ln_add<true><<<kM / 8, 256, 0, stream>>>(audio_src, TMP, ln1_g, ln1_b, AUDIO, AUDIO16);

  // ---- Stage C: vaa = msda(query=AUDIO, src=VIDEO) ; emits a_loc, a_aw
  msda_stage(AUDIO16, VIDEO16, audio_ref, video_ts, video_lsi, video_mask,
             o_aloc, o_aaw, VAA, VAA16);

  // ---- Stage D: aav = msda(query=VIDEO, src=AUDIO) ; emits v_loc, v_aw
  msda_stage(VIDEO16, AUDIO16, video_ref, audio_ts, audio_lsi, audio_mask,
             o_vloc, o_vaw, AAV, AAV16);

  // ---- FFNs (H16 aliases q16..SAMP16, all dead by now)
  ffn(VAA, VAA16, o_vaa);
  ffn(AAV, AAV16, o_aav);
}
